// ClusterLayer_50036368998425
// MI455X (gfx1250) — compile-verified, matrix operand order checked by reading, not run on hardware
//
#include <hip/hip_runtime.h>

// ---------------------------------------------------------------------------
// ClusterLayer for MI455X (gfx1250, wave32, WMMA)
//   d = ||x||^2 + ||w||^2 - 2 x.w^T ; cl = argmin_k d ; inertia = sum min d / B
//   xe = (2*sum_k ||w_k||^2 - || sum_k w_k ||^2) / K
// Cross term via v_wmma_f32_16x16x32_bf16 with bf16 hi/lo error compensation.
// v3: no VGPR spills — W chunks staged via async global->LDS DMA (ASYNCcnt),
//     argmin tracks score = w2 - 2*dot (x2 folded in at epilogue from L2),
//     w2 scalars prefetched per chunk. 96 static WMMAs, 4 indep chains/tile.
// ---------------------------------------------------------------------------

#define BATCH 131072
#define KCLUST 1024
#define DIM 128
#define USE_ASYNC_LDS 1

typedef __attribute__((ext_vector_type(16))) __bf16 v16bf;
typedef __attribute__((ext_vector_type(8)))  float  v8f;
typedef __attribute__((ext_vector_type(8)))  unsigned short ushort8;

union VB {
    ushort8 u2[2];
    v16bf   v;
    unsigned short us[16];
};

// round-to-nearest-even float -> bf16 hi, plus bf16 of the residual (lo)
__device__ __forceinline__ void bf16split(float f, unsigned short& h, unsigned short& l) {
    unsigned u = __float_as_uint(f);
    unsigned r = u + 0x7FFFu + ((u >> 16) & 1u);
    h = (unsigned short)(r >> 16);
    float hf = __uint_as_float(((unsigned)h) << 16);
    float lo = f - hf;
    unsigned v = __float_as_uint(lo);
    unsigned s = v + 0x7FFFu + ((v >> 16) & 1u);
    l = (unsigned short)(s >> 16);
}

#define WMMA_BF16(A, Bm, C) \
    __builtin_amdgcn_wmma_f32_16x16x32_bf16(false, (A), false, (Bm), (short)0, (C), false, false)

#if USE_ASYNC_LDS
// CDNA5 async DMA: global -> LDS, no data VGPRs, tracked by ASYNCcnt.
__device__ __forceinline__ void async_b128(unsigned lds_off, const void* gaddr) {
    asm volatile("global_load_async_to_lds_b128 %0, %1, off"
                 :: "v"(lds_off), "v"(gaddr) : "memory");
}
__device__ __forceinline__ void wait_async0() {
    asm volatile("s_wait_asynccnt 0x0" ::: "memory");
}
#endif

// Build the 16x32-bf16 A fragments (hi+lo) for one 16-row block.
__device__ __forceinline__ void build_A(const float* __restrict__ xrow, int half,
                                        v16bf* a_hi, v16bf* a_lo) {
    #pragma unroll
    for (int ks = 0; ks < 4; ++ks) {
        const int kb = 32 * ks + 8 * half;
        VB h, l;
        #pragma unroll
        for (int j = 0; j < 8; ++j) {
            bf16split(xrow[kb + j],      h.us[j],     l.us[j]);
            bf16split(xrow[kb + 16 + j], h.us[8 + j], l.us[8 + j]);
        }
        a_hi[ks] = h.v;
        a_lo[ks] = l.v;
    }
}

// ---------------------------------------------------------------------------
// Kernel 1: W -> bf16 hi/lo + per-row squared norms.  grid=K, 128 threads.
// ---------------------------------------------------------------------------
__global__ __launch_bounds__(128)
void prep_kernel(const float* __restrict__ W,
                 unsigned short* __restrict__ WH,
                 unsigned short* __restrict__ WL,
                 float* __restrict__ w2) {
    __shared__ float red4[4];
    const int k = blockIdx.x;
    const int d = threadIdx.x;
    float w = W[k * DIM + d];
    unsigned short h, l;
    bf16split(w, h, l);
    WH[k * DIM + d] = h;
    WL[k * DIM + d] = l;
    float p = w * w;
    #pragma unroll
    for (int m = 16; m >= 1; m >>= 1) p += __shfl_xor(p, m, 32);
    if ((d & 31) == 0) red4[d >> 5] = p;
    __syncthreads();
    if (d == 0) w2[k] = red4[0] + red4[1] + red4[2] + red4[3];
}

// ---------------------------------------------------------------------------
// Kernel 2: xe_loss = (2*sum(w2) - ||colsum(W)||^2) / K   (1 block, 128 thr)
// ---------------------------------------------------------------------------
__global__ __launch_bounds__(128)
void xe_kernel(const float* __restrict__ W,
               const float* __restrict__ w2,
               float* __restrict__ out) {
    __shared__ float red[128];
    const int d = threadIdx.x;
    float s = 0.f;
    for (int k = 0; k < KCLUST; ++k) s += W[k * DIM + d];
    float tw = 0.f;
    #pragma unroll
    for (int i = 0; i < KCLUST / DIM; ++i) tw += w2[d + DIM * i];

    red[d] = s * s;
    __syncthreads();
    for (int off = 64; off >= 1; off >>= 1) {
        if (d < off) red[d] += red[d + off];
        __syncthreads();
    }
    float ssq = red[0];
    __syncthreads();
    red[d] = tw;
    __syncthreads();
    for (int off = 64; off >= 1; off >>= 1) {
        if (d < off) red[d] += red[d + off];
        __syncthreads();
    }
    if (d == 0) out[1] = (2.f * red[0] - ssq) / (float)KCLUST;
}

// ---------------------------------------------------------------------------
// Kernel 3: fused GEMM (bf16 hi/lo WMMA) + argmin over score = w2 - 2*dot
// grid = B/256, 256 threads = 8 waves; wave owns 32 rows as two A-blocks.
// K in 16 chunks of 64 centroids; LDS ping-pong fed by async DMA.
// ---------------------------------------------------------------------------
__global__ __launch_bounds__(256)
void cluster_main(const float* __restrict__ X,
                  const unsigned short* __restrict__ WH,
                  const unsigned short* __restrict__ WL,
                  const float* __restrict__ w2,
                  float* __restrict__ out,
                  float* __restrict__ partials) {
    __shared__ unsigned short wh_s[2 * 64 * DIM];   // 32 KB (ping-pong hi)
    __shared__ unsigned short wl_s[2 * 64 * DIM];   // 32 KB (ping-pong lo)

    const int tid  = threadIdx.x;
    const int wave = tid >> 5;
    const int lid  = tid & 31;
    const int half = lid >> 4;
    const int l16  = lid & 15;
    const int mbase = blockIdx.x * 256 + wave * 32;

#if USE_ASYNC_LDS
    const unsigned whBase = (unsigned)(size_t)(void*)wh_s;   // low 32 bits = LDS offset
    const unsigned wlBase = (unsigned)(size_t)(void*)wl_s;
    // each thread DMAs 4 x b128 per array per chunk (16 KB / 256 threads / 16 B)
    auto issue_chunk = [&](int ch, int buf) {
        const char* gh = (const char*)WH + ch * 16384 + tid * 16;
        const char* gl = (const char*)WL + ch * 16384 + tid * 16;
        const unsigned lh = whBase + buf * 16384 + tid * 16;
        const unsigned ll = wlBase + buf * 16384 + tid * 16;
        #pragma unroll
        for (int i = 0; i < 4; ++i) {
            async_b128(lh + i * 4096, gh + i * 4096);
            async_b128(ll + i * 4096, gl + i * 4096);
        }
    };
    issue_chunk(0, 0);          // DMA of chunk 0 overlaps A-fragment build
#endif

    // ---- A fragments for two 16-row blocks ---------------------------------
    v16bf a_hi0[4], a_lo0[4], a_hi1[4], a_lo1[4];
    build_A(X + (size_t)(mbase      + l16) * DIM, half, a_hi0, a_lo0);
    build_A(X + (size_t)(mbase + 16 + l16) * DIM, half, a_hi1, a_lo1);

    float minv0[8], minv1[8];
    int   mini0[8], mini1[8];
    #pragma unroll
    for (int r = 0; r < 8; ++r) {
        minv0[r] = 3.4e38f; mini0[r] = 0;
        minv1[r] = 3.4e38f; mini1[r] = 0;
    }

    for (int ch = 0; ch < KCLUST / 64; ++ch) {
        const int buf = ch & 1;
        const unsigned short* bufh = wh_s + buf * 64 * DIM;
        const unsigned short* bufl = wl_s + buf * 64 * DIM;
#if USE_ASYNC_LDS
        wait_async0();                 // chunk `ch` landed in LDS buf
        __syncthreads();               // all waves' DMAs done; buf^1 free
        if (ch + 1 < KCLUST / 64) issue_chunk(ch + 1, buf ^ 1);
#else
        __syncthreads();               // everyone done reading buf
        {
            uint4* dh = (uint4*)(wh_s + buf * 64 * DIM);
            uint4* dl = (uint4*)(wl_s + buf * 64 * DIM);
            const uint4* gh = (const uint4*)(WH + ch * 64 * DIM);
            const uint4* gl = (const uint4*)(WL + ch * 64 * DIM);
            #pragma unroll
            for (int i = 0; i < 4; ++i) { dh[tid + 256 * i] = gh[tid + 256 * i]; dl[tid + 256 * i] = gl[tid + 256 * i]; }
        }
        __syncthreads();
#endif
        // prefetch this chunk's w2 scalars (one column class per lane)
        float w2n[4];
        #pragma unroll
        for (int nt = 0; nt < 4; ++nt) w2n[nt] = w2[ch * 64 + nt * 16 + l16];

        #pragma unroll
        for (int nt = 0; nt < 4; ++nt) {
            const int wr = nt * 16 + l16;
            v8f acc0h = {}, acc0l = {}, acc1h = {}, acc1l = {};
            #pragma unroll
            for (int ks = 0; ks < 4; ++ks) {
                const int kb = 32 * ks + 8 * half;
                VB bh, bl;
                bh.u2[0] = *(const ushort8*)(bufh + wr * DIM + kb);
                bh.u2[1] = *(const ushort8*)(bufh + wr * DIM + kb + 16);
                bl.u2[0] = *(const ushort8*)(bufl + wr * DIM + kb);
                bl.u2[1] = *(const ushort8*)(bufl + wr * DIM + kb + 16);
                acc0h = WMMA_BF16(a_hi0[ks], bh.v, acc0h);
                acc1h = WMMA_BF16(a_hi1[ks], bh.v, acc1h);
                acc0l = WMMA_BF16(a_lo0[ks], bh.v, acc0l);
                acc1l = WMMA_BF16(a_lo1[ks], bh.v, acc1l);
                acc0l = WMMA_BF16(a_hi0[ks], bl.v, acc0l);
                acc1l = WMMA_BF16(a_hi1[ks], bl.v, acc1l);
            }
            const int ng = ch * 64 + nt * 16 + l16;
            #pragma unroll
            for (int r = 0; r < 8; ++r) {
                float v0 = w2n[nt] - 2.f * (acc0h[r] + acc0l[r]);   // score (x2 added later)
                if (v0 < minv0[r]) { minv0[r] = v0; mini0[r] = ng; }
                float v1 = w2n[nt] - 2.f * (acc1h[r] + acc1l[r]);
                if (v1 < minv1[r]) { minv1[r] = v1; mini1[r] = ng; }
            }
        }
    }

    // ---- argmin across the 16 lanes of each half ---------------------------
    #pragma unroll
    for (int r = 0; r < 8; ++r) {
        #pragma unroll
        for (int m = 8; m >= 1; m >>= 1) {
            float ov0 = __shfl_xor(minv0[r], m, 32);
            int   oi0 = __shfl_xor(mini0[r], m, 32);
            if (ov0 < minv0[r] || (ov0 == minv0[r] && oi0 < mini0[r])) { minv0[r] = ov0; mini0[r] = oi0; }
            float ov1 = __shfl_xor(minv1[r], m, 32);
            int   oi1 = __shfl_xor(mini1[r], m, 32);
            if (ov1 < minv1[r] || (ov1 == minv1[r] && oi1 < mini1[r])) { minv1[r] = ov1; mini1[r] = oi1; }
        }
    }

    // ---- epilogue: ||x||^2 from L2-resident X, inertia = x2 + min score ----
    const float4* xr0 = (const float4*)(X + (size_t)(mbase      + l16) * DIM + 64 * half);
    const float4* xr1 = (const float4*)(X + (size_t)(mbase + 16 + l16) * DIM + 64 * half);
    float p0 = 0.f, p1 = 0.f;
    #pragma unroll
    for (int i = 0; i < 16; ++i) {
        float4 a = xr0[i], b = xr1[i];
        p0 += a.x * a.x + a.y * a.y + a.z * a.z + a.w * a.w;
        p1 += b.x * b.x + b.y * b.y + b.z * b.z + b.w * b.w;
    }
    float x2f0 = p0 + __shfl_xor(p0, 16, 32);
    float x2f1 = p1 + __shfl_xor(p1, 16, 32);

    float s = 0.f;
    #pragma unroll
    for (int r = 0; r < 8; ++r) {
        s += (__shfl(x2f0, r + 8 * half, 32) + minv0[r]);
        s += (__shfl(x2f1, r + 8 * half, 32) + minv1[r]);
    }
    float wsum = s + __shfl_xor(s, 16, 32);   // combine both halves' 16 rows

    __syncthreads();                          // done with LDS chunks; reuse space
    float* waveSums = (float*)wh_s;
    if (lid == 0) waveSums[wave] = wsum;
    if (l16 == 0) {
        #pragma unroll
        for (int r = 0; r < 8; ++r) {
            out[2 + mbase +      r + 8 * half] = (float)mini0[r];
            out[2 + mbase + 16 + r + 8 * half] = (float)mini1[r];
        }
    }
    __syncthreads();
    if (tid == 0) {
        float t = 0.f;
        #pragma unroll
        for (int i = 0; i < 8; ++i) t += waveSums[i];
        partials[blockIdx.x] = t;
    }
}

// ---------------------------------------------------------------------------
// Kernel 4: deterministic sum of per-WG inertia partials -> d_out[0]
// ---------------------------------------------------------------------------
__global__ __launch_bounds__(256)
void fin_kernel(const float* __restrict__ partials, float* __restrict__ out) {
    __shared__ float red[256];
    const int t = threadIdx.x;
    float s = 0.f;
    for (int i = t; i < BATCH / 256; i += 256) s += partials[i];
    red[t] = s;
    __syncthreads();
    for (int off = 128; off >= 1; off >>= 1) {
        if (t < off) red[t] += red[t + off];
        __syncthreads();
    }
    if (t == 0) out[0] = red[0] / (float)BATCH;
}

// ---------------------------------------------------------------------------
extern "C" void kernel_launch(void* const* d_in, const int* in_sizes, int n_in,
                              void* d_out, int out_size, void* d_ws, size_t ws_size,
                              hipStream_t stream) {
    const float* x  = (const float*)d_in[0];          // [B, D]
    const float* cw = (const float*)d_in[1];          // [K, D]
    float* out = (float*)d_out;                       // [inertia, xe, cl(B)]

    unsigned short* WH = (unsigned short*)d_ws;                  // K*D bf16 hi
    unsigned short* WL = WH + KCLUST * DIM;                      // K*D bf16 lo
    float* w2       = (float*)(WL + KCLUST * DIM);               // K norms
    float* partials = w2 + KCLUST;                               // B/256 sums

    prep_kernel<<<KCLUST, 128, 0, stream>>>(cw, WH, WL, w2);
    xe_kernel<<<1, 128, 0, stream>>>(cw, w2, out);
    cluster_main<<<BATCH / 256, 256, 0, stream>>>(x, WH, WL, w2, out, partials);
    fin_kernel<<<1, 256, 0, stream>>>(partials, out);
}
